// WaveletDeconvolutionModel_47450798686646
// MI455X (gfx1250) — compile-verified, matrix operand order checked by reading, not run on hardware
//
#include <hip/hip_runtime.h>
#include <hip/hip_bf16.h>

// ---------------------------------------------------------------------------
// Shapes
// ---------------------------------------------------------------------------
#define B_ROWS 4096
#define N_IN   10000
#define H_DIM  512
#define C_DIM  39
#define CL     10020      // 1256 + 1256 + 2505 + 5003
#define KP     10048      // CL padded up to a multiple of 32 (zero fill)
#define L1_LEN 5003
#define L2_LEN 2505
#define L3_LEN 1256
#define OFF_CA3 0
#define OFF_D3  1256
#define OFF_D2  2512
#define OFF_D1  5017

// ---------------------------------------------------------------------------
// Types for WMMA
// ---------------------------------------------------------------------------
typedef __bf16           v16bf __attribute__((ext_vector_type(16)));
typedef float            v8f   __attribute__((ext_vector_type(8)));
typedef unsigned short   v8u   __attribute__((ext_vector_type(8)));
typedef unsigned short   v16u  __attribute__((ext_vector_type(16)));

union Frag { v16u u; v16bf b; };

// float -> bf16 round-to-nearest-even
__device__ __forceinline__ unsigned short f2bf(float f) {
    unsigned int u = __float_as_uint(f);
    unsigned int r = u + 0x7FFFu + ((u >> 16) & 1u);
    return (unsigned short)(r >> 16);
}

// Wavelet filters: K_LO = _REC_LO ; K_HI[t] = _REC_LO[7-t] * (-1)^t
__device__ constexpr float KLO[8] = {
     0.23037781330885523f,  0.7148465705525415f,  0.6308807679295904f,
    -0.02798376941698385f, -0.18703481171888114f, 0.030841381835986965f,
     0.032883011666982945f, -0.010597401784997278f };
__device__ constexpr float KHI[8] = {
    -0.010597401784997278f, -0.032883011666982945f, 0.030841381835986965f,
     0.18703481171888114f,  -0.02798376941698385f, -0.6308807679295904f,
     0.7148465705525415f,   -0.23037781330885523f };

// ---------------------------------------------------------------------------
// Kernel 1: NaN mask + x*log1p(cov) + 3-level DWT (symmetric pad), bf16 out.
// One workgroup per row; row staged in LDS.
// xe[i] (i in [0, n+13)) maps to: i<6 -> x[5-i]; i-6<n -> x[i-6]; else x[2n-1-(i-6)]
// out[k] = sum_t xe[2k+t] * K[t]
// ---------------------------------------------------------------------------
__device__ __forceinline__ void dwt_level(const float* s, int n, int L,
                                          float* caLds, unsigned short* caBf,
                                          unsigned short* cdBf, int tid) {
    for (int k = tid; k < L; k += 256) {
        float lo = 0.f, hi = 0.f;
        const int base = 2 * k;
        if (base >= 6 && base <= n - 2) {            // interior fast path
            const float* p = s + (base - 6);
#pragma unroll
            for (int t = 0; t < 8; ++t) {
                float v = p[t];
                lo = fmaf(v, KLO[t], lo);
                hi = fmaf(v, KHI[t], hi);
            }
        } else {                                     // symmetric boundary
#pragma unroll
            for (int t = 0; t < 8; ++t) {
                int i = base + t, m;
                if (i < 6) m = 5 - i;
                else { int j = i - 6; m = (j < n) ? j : (2 * n - 1 - j); }
                float v = s[m];
                lo = fmaf(v, KLO[t], lo);
                hi = fmaf(v, KHI[t], hi);
            }
        }
        if (caLds) caLds[k] = lo; else caBf[k] = f2bf(lo);
        cdBf[k] = f2bf(hi);
    }
}

__global__ __launch_bounds__(256)
void dwt_kernel(const float* __restrict__ X, const float* __restrict__ Cov,
                unsigned short* __restrict__ F) {
    __shared__ float sX[N_IN];     // 40000 B
    __shared__ float sC[L1_LEN];   // 20012 B
    const int b   = blockIdx.x;
    const int tid = threadIdx.x;

    const float* xr = X   + (size_t)b * N_IN;
    const float* cr = Cov + (size_t)b * N_IN;
    for (int i = tid; i < N_IN; i += 256) {
        float x = xr[i];
        float c = cr[i];
        if (x != x) { x = 0.f; c = 0.f; }            // NaN mask (both)
        sX[i] = x * log1pf(c);
    }
    __syncthreads();

    unsigned short* fr = F + (size_t)b * KP;
    dwt_level(sX, N_IN,   L1_LEN, sC, nullptr,      fr + OFF_D1, tid);  // L1
    __syncthreads();
    dwt_level(sC, L1_LEN, L2_LEN, sX, nullptr,      fr + OFF_D2, tid);  // L2
    __syncthreads();
    dwt_level(sX, L2_LEN, L3_LEN, nullptr, fr + OFF_CA3, fr + OFF_D3, tid); // L3
    for (int k = CL + tid; k < KP; k += 256) fr[k] = 0;                 // pad
}

// ---------------------------------------------------------------------------
// Kernel 2: W1 (512 x 10020 f32) -> bf16, K padded to KP with zeros.
// ---------------------------------------------------------------------------
__global__ __launch_bounds__(256)
void convw1_kernel(const float* __restrict__ W1, unsigned short* __restrict__ W1b) {
    size_t idx = (size_t)blockIdx.x * 256 + threadIdx.x;
    if (idx >= (size_t)H_DIM * KP) return;
    size_t n = idx / KP, k = idx % KP;
    W1b[idx] = (k < CL) ? f2bf(W1[n * CL + k]) : (unsigned short)0;
}

// ---------------------------------------------------------------------------
// Kernel 3: GEMM  H = feats(4096 x KP, bf16) x W1b^T(KP x 512, bf16) -> f32
// Block tile 128(M) x 64(N), 4 waves; wave tile 32(M) x 64(N):
//   2 M-frags x 4 N-frags -> 8 WMMA 16x16x32 accumulators per wave,
//   12 b128 loads : 8 WMMAs per 32-K step (vs 2:1 with a 32x32 tile).
// A fragment per lane (16-bit 16x32 ISA layout): lanes 0-15 hold K {0..7,16..23},
// lanes 16-31 hold K {8..15,24..31}; B uses the mirrored layout with lane = N.
// ---------------------------------------------------------------------------
__global__ __launch_bounds__(128)
void gemm1_kernel(const unsigned short* __restrict__ A,
                  const unsigned short* __restrict__ Bm,
                  float* __restrict__ H) {
    const int tid   = threadIdx.x;
    const int wave  = tid >> 5;
    const int lane  = tid & 31;
    const int laneM = lane & 15;
    const int halfK = (lane >> 4) * 8;

    const int mBase = blockIdx.x * 128 + wave * 32;   // wave owns 32 M-rows
    const int nBase = blockIdx.y * 64;                // block owns 64 N-cols

    const unsigned short* aP[2];
    const unsigned short* bP[4];
#pragma unroll
    for (int i = 0; i < 2; ++i)
        aP[i] = A  + (size_t)(mBase + i * 16 + laneM) * KP + halfK;
#pragma unroll
    for (int j = 0; j < 4; ++j)
        bP[j] = Bm + (size_t)(nBase + j * 16 + laneM) * KP + halfK;

    v8f acc[2][4] = {};

    for (int k = 0; k < KP; k += 32) {
        // A streams from HBM: prefetch into WGP cache ~2KB ahead.
        __builtin_prefetch(aP[0] + k + 1024, 0, 3);
        Frag a[2], b[4];
#pragma unroll
        for (int i = 0; i < 2; ++i) {
            v8u lo = *(const v8u*)(aP[i] + k);
            v8u hi = *(const v8u*)(aP[i] + k + 16);
            a[i].u = __builtin_shufflevector(lo, hi,
                     0,1,2,3,4,5,6,7,8,9,10,11,12,13,14,15);
        }
#pragma unroll
        for (int j = 0; j < 4; ++j) {
            v8u lo = *(const v8u*)(bP[j] + k);
            v8u hi = *(const v8u*)(bP[j] + k + 16);
            b[j].u = __builtin_shufflevector(lo, hi,
                     0,1,2,3,4,5,6,7,8,9,10,11,12,13,14,15);
        }
#pragma unroll
        for (int i = 0; i < 2; ++i)
#pragma unroll
            for (int j = 0; j < 4; ++j)
                acc[i][j] = __builtin_amdgcn_wmma_f32_16x16x32_bf16(
                    false, a[i].b, false, b[j].b,
                    (short)0, acc[i][j], false, false);
    }

    // C/D layout: VGPR r -> (M=r, N=lane) for lanes 0-15, (M=r+8, N=lane-16) else
    const int rowHalf = (lane >> 4) * 8;
#pragma unroll
    for (int i = 0; i < 2; ++i)
#pragma unroll
        for (int j = 0; j < 4; ++j)
#pragma unroll
            for (int r = 0; r < 8; ++r) {
                int row = mBase + i * 16 + r + rowHalf;
                int col = nBase + j * 16 + laneM;
                H[(size_t)row * H_DIM + col] = acc[i][j][r];
            }
}

// ---------------------------------------------------------------------------
// Kernel 4: per-row LayerNorm + ReLU + (512->39) GEMM + softmax.
// One 256-thread block per row.
// ---------------------------------------------------------------------------
__global__ __launch_bounds__(256)
void finalize_kernel(const float* __restrict__ Hb, const float* __restrict__ b1,
                     const float* __restrict__ gamma, const float* __restrict__ beta,
                     const float* __restrict__ W2, const float* __restrict__ b2,
                     float* __restrict__ out) {
    __shared__ float sh[H_DIM];
    __shared__ float rbuf[256];
    __shared__ float slog[C_DIM];
    const int b = blockIdx.x, tid = threadIdx.x;

    float v0 = Hb[(size_t)b * H_DIM + tid]       + b1[tid];
    float v1 = Hb[(size_t)b * H_DIM + tid + 256] + b1[tid + 256];

    rbuf[tid] = v0 + v1;
    __syncthreads();
    for (int s = 128; s > 0; s >>= 1) {
        if (tid < s) rbuf[tid] += rbuf[tid + s];
        __syncthreads();
    }
    const float mean = rbuf[0] * (1.f / H_DIM);
    __syncthreads();

    const float d0 = v0 - mean, d1 = v1 - mean;
    rbuf[tid] = d0 * d0 + d1 * d1;
    __syncthreads();
    for (int s = 128; s > 0; s >>= 1) {
        if (tid < s) rbuf[tid] += rbuf[tid + s];
        __syncthreads();
    }
    const float rstd = rsqrtf(rbuf[0] * (1.f / H_DIM) + 1e-5f);
    __syncthreads();

    sh[tid]       = fmaxf(fmaf(d0 * rstd, gamma[tid],       beta[tid]),       0.f);
    sh[tid + 256] = fmaxf(fmaf(d1 * rstd, gamma[tid + 256], beta[tid + 256]), 0.f);
    __syncthreads();

    if (tid < C_DIM) {
        const float* w = W2 + (size_t)tid * H_DIM;
        float s = b2[tid];
#pragma unroll 4
        for (int k = 0; k < H_DIM; ++k) s = fmaf(sh[k], w[k], s);
        slog[tid] = s;
    }
    __syncthreads();

    if (tid < C_DIM) {
        float mx = -3.4e38f;
        for (int c = 0; c < C_DIM; ++c) mx = fmaxf(mx, slog[c]);
        float den = 0.f;
        for (int c = 0; c < C_DIM; ++c) den += __expf(slog[c] - mx);
        out[(size_t)b * C_DIM + tid] = __expf(slog[tid] - mx) / den;
    }
}

// ---------------------------------------------------------------------------
// Launch
// ---------------------------------------------------------------------------
extern "C" void kernel_launch(void* const* d_in, const int* in_sizes, int n_in,
                              void* d_out, int out_size, void* d_ws, size_t ws_size,
                              hipStream_t stream) {
    const float* X     = (const float*)d_in[0];
    const float* cov   = (const float*)d_in[1];
    const float* W1    = (const float*)d_in[2];
    const float* b1    = (const float*)d_in[3];
    const float* gamma = (const float*)d_in[4];
    const float* beta  = (const float*)d_in[5];
    const float* W2    = (const float*)d_in[6];
    const float* b2    = (const float*)d_in[7];
    float* out = (float*)d_out;

    // Workspace layout: feats bf16 [4096*KP] | W1b bf16 [512*KP] | H f32 [4096*512]
    unsigned short* feats = (unsigned short*)d_ws;
    unsigned short* w1b   = feats + (size_t)B_ROWS * KP;
    float*          Hbuf  = (float*)(w1b + (size_t)H_DIM * KP);

    dwt_kernel<<<dim3(B_ROWS), dim3(256), 0, stream>>>(X, cov, feats);

    size_t w1Elems = (size_t)H_DIM * KP;
    convw1_kernel<<<dim3((unsigned)((w1Elems + 255) / 256)), dim3(256), 0, stream>>>(W1, w1b);

    gemm1_kernel<<<dim3(B_ROWS / 128, H_DIM / 64), dim3(128), 0, stream>>>(feats, w1b, Hbuf);

    finalize_kernel<<<dim3(B_ROWS), dim3(256), 0, stream>>>(Hbuf, b1, gamma, beta, W2, b2, out);
}
